// MlpwithSOMModuleCosSimilarity_32452772889033
// MI455X (gfx1250) — matrix operand order repaired and checked by reading, not compiled, not run on hardware
//
#include <hip/hip_runtime.h>

typedef __attribute__((ext_vector_type(16))) _Float16 v16h;
typedef __attribute__((ext_vector_type(8)))  float    v8f;

#define B_   8
#define L_   2048
#define D_   128
#define NTILES2 (L_ / 32)         // 64 double-column-tile iterations per batch
#define ROWTILES (B_ * L_ / 16)   // 1024 row tiles total
#define AK_WAVES 4                // waves per argmax block (each owns 2 M-tiles)

struct alignas(16) H8 { _Float16 h[8]; };
struct alignas(8)  H4 { _Float16 h[4]; };

// ---------------------------------------------------------------------------
// A-matrix fragment (16-bit, 16x32; this code's k order, not the ISA table order):
//   lanes 0-15: K = kb+[0..15]; lanes 16-31: K = kb+[16..31] (contiguous, matches B)
__device__ inline v16h load_afrag(const _Float16* rowptr, int kb, int lane) {
  int s = (lane >> 4) << 4;                 // 0 or 16
  const H8* p0 = (const H8*)(rowptr + kb + s);
  const H8* p1 = (const H8*)(rowptr + kb + 8 + s);
  v16h a;
#pragma unroll
  for (int i = 0; i < 8; ++i) { a[i] = p0->h[i]; a[8 + i] = p1->h[i]; }
  return a;
}

// B-matrix fragment (16-bit, 32x16): lane = N (mod 16); lanes 0-15 hold K=kb+0..15,
// lanes 16-31 hold K=kb+16..31, 16 contiguous halfs along K per lane.
__device__ inline v16h load_bfrag(const _Float16* rowptr, int kb, int lane) {
  return *(const v16h*)(rowptr + kb + ((lane >> 4) << 4));
}

__device__ inline v8f wmma_f16(v16h a, v16h b, v8f c) {
  return __builtin_amdgcn_wmma_f32_16x16x32_f16(false, a, false, b, (short)0, c,
                                                false, false);
}

// Async global -> LDS copy, 16 bytes per lane (GLOBAL_LOAD_ASYNC_TO_LDS_B128,
// GV mode: 64-bit global address in VGPR pair, LDS byte offset in VDST VGPR).
__device__ inline void async_copy_b128(unsigned int lds_off, const void* gaddr) {
  asm volatile("global_load_async_to_lds_b128 %0, %1, off"
               :: "v"(lds_off), "v"((unsigned long long)(uintptr_t)gaddr)
               : "memory");
}

// ---------------------------------------------------------------------------
// Kernel 1: L2-normalize rows of context, write f16 cn / en.
__global__ __launch_bounds__(256) void norm_kernel(const float* __restrict__ ctx,
                                                   _Float16* __restrict__ cn16,
                                                   _Float16* __restrict__ en16) {
  int wave = threadIdx.x >> 5;
  int lane = threadIdx.x & 31;
  int row = blockIdx.x * 8 + wave;                 // 0 .. B*2*L-1
  const float* src = ctx + (size_t)row * D_ + lane * 4;
  float4 v = *(const float4*)src;
  float ss = v.x * v.x + v.y * v.y + v.z * v.z + v.w * v.w;
#pragma unroll
  for (int mask = 16; mask >= 1; mask >>= 1) ss += __shfl_xor(ss, mask, 32);
  float scale = 1.0f / fmaxf(sqrtf(ss), 1e-8f);

  int b   = row >> 12;        // / (2*L)
  int rem = row & 4095;
  int w   = rem >> 11;        // 0 = ctx, 1 = ent
  int l   = rem & 2047;
  _Float16* dst = (w == 0 ? cn16 : en16) + ((size_t)(b * L_ + l)) * D_ + lane * 4;
  H4 h;
  h.h[0] = (_Float16)(v.x * scale);
  h.h[1] = (_Float16)(v.y * scale);
  h.h[2] = (_Float16)(v.z * scale);
  h.h[3] = (_Float16)(v.w * scale);
  *(H4*)dst = h;
}

// ---------------------------------------------------------------------------
// Kernel 2: pre-shuffle W1 (f32 row-major DxD) into B-fragment layout (f16).
__global__ __launch_bounds__(256) void w1frag_kernel(const float* __restrict__ W1,
                                                     _Float16* __restrict__ w1frag) {
  int tid = blockIdx.x * 256 + threadIdx.x;
  if (tid >= 1024) return;
  int t   = tid & 31;
  int fid = tid >> 5;
  int jt  = fid >> 2;
  int kc  = fid & 3;
  int j   = jt * 16 + (t & 15);
  int kb  = kc * 32 + ((t >> 4) << 4);
  _Float16* dst = w1frag + (size_t)tid * 16;
#pragma unroll
  for (int i = 0; i < 16; ++i) dst[i] = (_Float16)W1[(size_t)(kb + i) * D_ + j];
}

// ---------------------------------------------------------------------------
// Kernel 3: fused sim = cn @ en^T + running argmax per row.
// Block = 4 waves; each wave owns TWO 16-row M-tiles (T0, T1 = T0+4 tiles).
// Each iteration stages a 32x128 en tile (8 KB) in LDS via async copies
// (double-buffered); every B-fragment read from LDS feeds two WMMAs.
__global__ __launch_bounds__(AK_WAVES * 32) void argmax_kernel(
    const _Float16* __restrict__ cn16, const _Float16* __restrict__ en16,
    int* __restrict__ idxOut) {
  __shared__ __align__(32) _Float16 smem[2][32 * D_];   // 2 x 8 KB

  int tid  = threadIdx.x;
  int wave = tid >> 5;
  int lane = tid & 31;
  int m    = lane & 15;
  int T0   = blockIdx.x * 8 + wave;           // first M-tile
  int T1   = T0 + AK_WAVES;                   // second M-tile
  int b    = blockIdx.x >> 4;                 // 16 blocks (128 tiles) per batch

  // Per-wave A fragments for both M-tiles, resident for the whole N loop.
  const _Float16* ar0 = cn16 + (size_t)(T0 * 16 + m) * D_;
  const _Float16* ar1 = cn16 + (size_t)(T1 * 16 + m) * D_;
  v16h a0[4], a1[4];
#pragma unroll
  for (int kc = 0; kc < 4; ++kc) {
    a0[kc] = load_afrag(ar0, kc * 32, lane);
    a1[kc] = load_afrag(ar1, kc * 32, lane);
  }

  const _Float16* enb = en16 + (size_t)b * L_ * D_;

  // Copy mapping: 128 threads x 64 B (4 x b128) per 8 KB tile.
  int hoff = tid * 32;                        // half-offset within 32x128 tile
  unsigned int ldsA = (unsigned int)(uintptr_t)(&smem[0][0]) + hoff * 2;
  unsigned int ldsB = (unsigned int)(uintptr_t)(&smem[1][0]) + hoff * 2;

  float best0[8], best1[8];
  int bidx0[8], bidx1[8];
#pragma unroll
  for (int r = 0; r < 8; ++r) {
    best0[r] = -3.4e38f; bidx0[r] = 0;
    best1[r] = -3.4e38f; bidx1[r] = 0;
  }

  // Prime tile 0 into buffer 0.
  {
    const _Float16* g = enb + hoff;
#pragma unroll
    for (int q = 0; q < 4; ++q) async_copy_b128(ldsA + q * 16, g + q * 8);
  }

  for (int nt2 = 0; nt2 < NTILES2; ++nt2) {
    int buf = nt2 & 1;
    if (nt2 + 1 < NTILES2) {
      const _Float16* g = enb + (size_t)(nt2 + 1) * 32 * D_ + hoff;
      unsigned int l = (buf ? ldsA : ldsB);
#pragma unroll
      for (int q = 0; q < 4; ++q) async_copy_b128(l + q * 16, g + q * 8);
      asm volatile("s_wait_asynccnt 0x4" ::: "memory");  // tile nt2 done
    } else {
      asm volatile("s_wait_asynccnt 0x0" ::: "memory");
    }
    __syncthreads();            // tile nt2 fully visible in LDS to all waves

#pragma unroll
    for (int u = 0; u < 2; ++u) {                 // two 16-column sub-tiles
      const _Float16* bt = &smem[buf][0] + (size_t)(u * 16 + m) * D_;
      v16h b0 = load_bfrag(bt, 0, lane);
      v16h b1 = load_bfrag(bt, 32, lane);
      v16h b2 = load_bfrag(bt, 64, lane);
      v16h b3 = load_bfrag(bt, 96, lane);
      v8f c0 = {}, c1 = {};
      c0 = wmma_f16(a0[0], b0, c0);  c1 = wmma_f16(a1[0], b0, c1);
      c0 = wmma_f16(a0[1], b1, c0);  c1 = wmma_f16(a1[1], b1, c1);
      c0 = wmma_f16(a0[2], b2, c0);  c1 = wmma_f16(a1[2], b2, c1);
      c0 = wmma_f16(a0[3], b3, c0);  c1 = wmma_f16(a1[3], b3, c1);

      int nidx = nt2 * 32 + u * 16 + m;           // N column this lane owns
#pragma unroll
      for (int r = 0; r < 8; ++r) {
        float v0 = c0[r];
        if (v0 > best0[r]) { best0[r] = v0; bidx0[r] = nidx; }
        float v1 = c1[r];
        if (v1 > best1[r]) { best1[r] = v1; bidx1[r] = nidx; }
      }
    }
    __syncthreads();            // all waves done reading buf before reuse
  }

  // argmax across the 16 lanes sharing each row (masks < 16 stay in-group)
#pragma unroll
  for (int mask = 1; mask <= 8; mask <<= 1) {
#pragma unroll
    for (int r = 0; r < 8; ++r) {
      float ov = __shfl_xor(best0[r], mask, 32);
      int   oi = __shfl_xor(bidx0[r], mask, 32);
      if (ov > best0[r] || (ov == best0[r] && oi < bidx0[r])) {
        best0[r] = ov; bidx0[r] = oi;
      }
      float ow = __shfl_xor(best1[r], mask, 32);
      int   oj = __shfl_xor(bidx1[r], mask, 32);
      if (ow > best1[r] || (ow == best1[r] && oj < bidx1[r])) {
        best1[r] = ow; bidx1[r] = oj;
      }
    }
  }

  if (m == 0) {
    int half = (lane >> 4) << 3;   // rows 0..7 (lane 0) / 8..15 (lane 16)
#pragma unroll
    for (int r = 0; r < 8; ++r) {
      idxOut[T0 * 16 + half + r] = bidx0[r];
      idxOut[T1 * 16 + half + r] = bidx1[r];
    }
  }
}

// ---------------------------------------------------------------------------
// Kernel 4: gather matched rows, MLP via WMMA (cn + matched fused: each W1
// fragment load feeds two WMMAs), relu, dot with W2, pair-sum, store.
__global__ __launch_bounds__(32) void mlp_kernel(const _Float16* __restrict__ cn16,
                                                 const _Float16* __restrict__ en16,
                                                 const int* __restrict__ idxArr,
                                                 const _Float16* __restrict__ w1frag,
                                                 const float* __restrict__ b1,
                                                 const float* __restrict__ W2,
                                                 const float* __restrict__ b2,
                                                 float* __restrict__ out) {
  int T = blockIdx.x;
  int lane = threadIdx.x;
  int m = lane & 15;
  int g = T * 16 + m;           // global row for this lane's A fragment
  int b = T >> 7;

  // A fragments: cn row (a[0..3]) and matched en row (a[4..7]).
  const _Float16* ar0 = cn16 + (size_t)g * D_;
  int mi = idxArr[g];
  const _Float16* ar1 = en16 + ((size_t)b * L_ + mi) * D_;
  v16h a[8];
#pragma unroll
  for (int kc = 0; kc < 4; ++kc) {
    a[kc]     = load_afrag(ar0, kc * 32, lane);
    a[4 + kc] = load_afrag(ar1, kc * 32, lane);
  }

  float s[8];
#pragma unroll
  for (int r = 0; r < 8; ++r) s[r] = 0.0f;

#pragma unroll
  for (int jt = 0; jt < 8; ++jt) {
    int j = jt * 16 + m;
    float b1j = b1[j];
    float w2j = W2[j];
    v8f c0, c1;
#pragma unroll
    for (int r = 0; r < 8; ++r) { c0[r] = b1j; c1[r] = b1j; }   // + b1
#pragma unroll
    for (int kc = 0; kc < 4; ++kc) {
      v16h bw = *(const v16h*)(w1frag + ((size_t)((jt * 4 + kc) * 32 + lane) << 4));
      c0 = wmma_f16(a[kc], bw, c0);
      c1 = wmma_f16(a[4 + kc], bw, c1);
    }
#pragma unroll
    for (int r = 0; r < 8; ++r)
      s[r] += (fmaxf(c0[r], 0.0f) + fmaxf(c1[r], 0.0f)) * w2j;
  }

  // row sum across the 16 lanes in each group
#pragma unroll
  for (int mask = 1; mask <= 8; mask <<= 1) {
#pragma unroll
    for (int r = 0; r < 8; ++r) s[r] += __shfl_xor(s[r], mask, 32);
  }

  if (m == 0) {
    float bb = 2.0f * b2[0];
    int base = T * 16 + ((lane >> 4) << 3);
#pragma unroll
    for (int r = 0; r < 8; ++r) out[base + r] = s[r] + bb;
  }
}

// ---------------------------------------------------------------------------
extern "C" void kernel_launch(void* const* d_in, const int* in_sizes, int n_in,
                              void* d_out, int out_size, void* d_ws, size_t ws_size,
                              hipStream_t stream) {
  const float* ctx = (const float*)d_in[0];   // (B,2,L,D) f32
  const float* W1  = (const float*)d_in[1];   // (D,D)
  const float* b1  = (const float*)d_in[2];   // (D,)
  const float* W2  = (const float*)d_in[3];   // (D,1)
  const float* b2  = (const float*)d_in[4];   // (1,)
  float* out = (float*)d_out;                 // (B*L,)

  char* ws = (char*)d_ws;
  const size_t cn_bytes = (size_t)B_ * L_ * D_ * sizeof(_Float16);   // 4 MB
  _Float16* cn16   = (_Float16*)(ws);
  _Float16* en16   = (_Float16*)(ws + cn_bytes);
  int*      idxArr = (int*)(ws + 2 * cn_bytes);
  _Float16* w1frag = (_Float16*)(ws + 2 * cn_bytes + (size_t)B_ * L_ * sizeof(int));

  norm_kernel<<<(B_ * 2 * L_) / 8, 256, 0, stream>>>(ctx, cn16, en16);
  w1frag_kernel<<<4, 256, 0, stream>>>(W1, w1frag);
  argmax_kernel<<<ROWTILES / 8, AK_WAVES * 32, 0, stream>>>(cn16, en16, idxArr);
  mlp_kernel<<<ROWTILES, 32, 0, stream>>>(cn16, en16, idxArr, w1frag, b1, W2, b2, out);
}
